// _CausalLayerCached_32633161515009
// MI455X (gfx1250) — compile-verified, matrix operand order checked by reading, not run on hardware
//
#include <hip/hip_runtime.h>

// ---------------------------------------------------------------------------
// Types for CDNA5 WMMA (wave32)
// ---------------------------------------------------------------------------
typedef __bf16 v16bf __attribute__((ext_vector_type(16)));
typedef float  v8f   __attribute__((ext_vector_type(8)));
typedef unsigned int u32x4 __attribute__((ext_vector_type(4)));
typedef int i32x4 __attribute__((ext_vector_type(4)));
typedef int i32x8 __attribute__((ext_vector_type(8)));

__device__ __forceinline__ v8f wmma_bf16(v16bf a, v16bf b, v8f c) {
  // D = A(16x32 bf16) * B(32x16 bf16) + C(16x16 f32)
  return __builtin_amdgcn_wmma_f32_16x16x32_bf16(
      false, a, false, b, (short)0, c, false, false);
}

#if defined(__has_builtin)
#if __has_builtin(__builtin_amdgcn_tensor_load_to_lds) && \
    __has_builtin(__builtin_amdgcn_s_wait_tensorcnt)
#define USE_TDM 1
#endif
#endif

#ifdef USE_TDM
// 2D tile load Global->LDS via Tensor Data Mover (bf16 payload).
__device__ __forceinline__ void tdm_load_2d(unsigned lds_off, const __bf16* gaddr,
                                            unsigned dim0, unsigned dim1,
                                            unsigned tile0, unsigned tile1,
                                            unsigned stride0,
                                            unsigned pad_int_code,
                                            unsigned pad_amt_code) {
  unsigned long long ga = (unsigned long long)(uintptr_t)gaddr;
  u32x4 g0;
  g0[0] = 1u;                                       // count=1, user descriptor
  g0[1] = lds_off;                                  // lds_addr
  g0[2] = (unsigned)(ga & 0xFFFFFFFFu);             // global_addr[31:0]
  g0[3] = (unsigned)((ga >> 32) & 0x01FFFFFFu)      // global_addr[56:32]
          | (2u << 30);                             // type = 2 (image)
  i32x8 g1;
  g1[0] = (int)((1u << 16)                          // data_size = 2 bytes
                | (1u << 20)                        // pad_enable
                | (pad_int_code << 22)
                | (pad_amt_code << 25));
  g1[1] = (int)((dim0 & 0xFFFFu) << 16);            // tensor_dim0[15:0]
  g1[2] = (int)((dim0 >> 16) | ((dim1 & 0xFFFFu) << 16));
  g1[3] = (int)((dim1 >> 16) | (tile0 << 16));      // tile_dim0
  g1[4] = (int)(tile1 & 0xFFFFu);                   // tile_dim1 (tile_dim2=0)
  g1[5] = (int)stride0;                             // tensor_dim0_stride[31:0]
  g1[6] = 0;
  g1[7] = 0;
  i32x4 z4 = {0, 0, 0, 0};
#if defined(__clang_major__) && (__clang_major__ >= 23)
  i32x8 z8 = {0, 0, 0, 0, 0, 0, 0, 0};
  __builtin_amdgcn_tensor_load_to_lds(g0, g1, z4, z4, z8, 0);
#else
  __builtin_amdgcn_tensor_load_to_lds(g0, g1, z4, z4, 0);
#endif
}
#endif  // USE_TDM

#define D_MODEL 1024
#define NHEAD   16
#define HEAD_DIM 64
#define DIM_FF  4096
#define BATCH   4
#define TC      1024
#define TP      1024
#define TKMAX   2048
#define NTOK    (BATCH * TC)   // 4096

// ---------------------------------------------------------------------------
// fp32 [K,N] -> bf16 transposed [N,K] (one-time weight prep: GEMM B tiles
// become k-contiguous and stageable via 128-bit copies / TDM)
// ---------------------------------------------------------------------------
__global__ __launch_bounds__(256) void k_cvt_transpose(const float* __restrict__ src,
                                                       __bf16* __restrict__ dst,
                                                       int K, int N) {
  int i = blockIdx.x * 256 + threadIdx.x;
  if (i >= K * N) return;
  int n = i / K, k = i - n * K;
  dst[i] = (__bf16)src[(size_t)k * N + n];
}

// ---------------------------------------------------------------------------
// LayerNorm row kernel: one block (256 thr) per row of D_MODEL, bf16 out
// ---------------------------------------------------------------------------
__global__ __launch_bounds__(256) void k_layernorm(const float* __restrict__ x,
                                                   const float* __restrict__ g,
                                                   const float* __restrict__ b,
                                                   __bf16* __restrict__ out) {
  const int row = blockIdx.x;
  const int tid = threadIdx.x;
  const int lane = tid & 31, wave = tid >> 5;
  const float* xr = x + (size_t)row * D_MODEL;
  __shared__ float rs[8], rs2[8];

  float s = 0.f, s2 = 0.f;
  for (int i = tid; i < D_MODEL; i += 256) {
    float v = xr[i];
    s += v; s2 += v * v;
  }
  #pragma unroll
  for (int off = 16; off >= 1; off >>= 1) {
    s  += __shfl_xor(s,  off, 32);
    s2 += __shfl_xor(s2, off, 32);
  }
  if (lane == 0) { rs[wave] = s; rs2[wave] = s2; }
  __syncthreads();
  if (tid == 0) {
    float S = 0.f, S2 = 0.f;
    #pragma unroll
    for (int i = 0; i < 8; ++i) { S += rs[i]; S2 += rs2[i]; }
    rs[0] = S; rs2[0] = S2;
  }
  __syncthreads();
  const float mu = rs[0] * (1.f / D_MODEL);
  const float var = rs2[0] * (1.f / D_MODEL) - mu * mu;
  const float rstd = rsqrtf(var + 1e-5f);
  for (int i = tid; i < D_MODEL; i += 256) {
    out[(size_t)row * D_MODEL + i] = (__bf16)((xr[i] - mu) * rstd * g[i] + b[i]);
  }
}

// ---------------------------------------------------------------------------
// Copy past KV into output cache (fp32) + bf16 shadows for attention.
// K shadow: [B,H,Tk,Dh] row-major; V shadow: TRANSPOSED [B,H,Dh,Tk].
// ---------------------------------------------------------------------------
__global__ __launch_bounds__(256) void k_kvcopy(const float* __restrict__ Kp,
                                                const float* __restrict__ Vp,
                                                float* __restrict__ Kout,
                                                float* __restrict__ Vout,
                                                __bf16* __restrict__ Kbf,
                                                __bf16* __restrict__ Vtbf) {
  int i = blockIdx.x * 256 + threadIdx.x;
  const int n = BATCH * NHEAD * TP * HEAD_DIM;
  if (i >= n) return;
  int bh  = i / (TP * HEAD_DIM);
  int rem = i - bh * (TP * HEAD_DIM);
  int t = rem >> 6, d = rem & 63;
  size_t dst = (size_t)bh * TKMAX * HEAD_DIM + (size_t)t * HEAD_DIM + d;
  float kv = Kp[i], vv = Vp[i];
  Kout[dst] = kv; Vout[dst] = vv;
  Kbf[dst] = (__bf16)kv;
  Vtbf[(size_t)bh * HEAD_DIM * TKMAX + (size_t)d * TKMAX + t] = (__bf16)vv;
}

// ---------------------------------------------------------------------------
// Tiled WMMA GEMM: C[M,N] = A[M,K](bf16) * Bt[N,K](bf16, pre-transposed)
//   block tile 128x128, K-step 32, 256 threads = 8 waves,
//   wave grid 4(M) x 2(N); each wave: 2x4 accumulators of 16x16.
//   TDM path: double-buffered pipelined tile loads (s_wait_tensorcnt 2).
// MODE 0: QKV scatter  MODE 1: bias+residual->f32  MODE 2: bias+GELU->bf16
// ---------------------------------------------------------------------------
#define GBM 128
#define GBN 128
#define GBK 32
#define LDSK 40                                     // row stride (elements)
#define SA_BYTES (GBM * LDSK * 2)                   // 10240
#define GEMM_BUF_BYTES (2 * SA_BYTES)               // A + B per buffer

template <int MODE>
__global__ __launch_bounds__(256) void k_gemm_bf16(
    const __bf16* __restrict__ A, const __bf16* __restrict__ Bt,
    int M, int N, int K,
    const float* __restrict__ bias,
    float* __restrict__ Koutf, float* __restrict__ Voutf,
    __bf16* __restrict__ Kbf, __bf16* __restrict__ Vtbf, __bf16* __restrict__ Qbf,
    const float* __restrict__ resid, float* __restrict__ outf,
    __bf16* __restrict__ outbf) {
  __shared__ __align__(16) char smem[2 * GEMM_BUF_BYTES];   // double buffered

  const int tid = threadIdx.x;
  const int wave = tid >> 5, lane = tid & 31;
  const int mtiles = M / GBM;
  const int bm = blockIdx.x % mtiles;
  const int bn = blockIdx.x / mtiles;
  const int wm = wave & 3, wn = wave >> 2;
  const int row0 = bm * GBM + wm * 32;
  const int col0 = bn * GBN + wn * 64;

  const __bf16* Abase = A  + (size_t)(bm * GBM) * K;
  const __bf16* Bbase = Bt + (size_t)(bn * GBN) * K;

  v8f acc[2][4];
  #pragma unroll
  for (int i = 0; i < 2; ++i)
    #pragma unroll
    for (int j = 0; j < 4; ++j) acc[i][j] = (v8f)(0.f);

  const int lr = lane & 15;
  const int lh = (lane >> 4) * 16;
  const int KT = K / GBK;

#ifdef USE_TDM
  if (wave == 0) {
    tdm_load_2d(0,        Abase, K, M, GBK, GBM, K, 3, 3);
    tdm_load_2d(SA_BYTES, Bbase, K, N, GBK, GBN, K, 3, 3);
  }
#endif

  for (int kt = 0; kt < KT; ++kt) {
    const int k0 = kt * GBK;
    const int cur = kt & 1;
    __bf16 (*cA)[LDSK] = (__bf16(*)[LDSK])(smem + cur * GEMM_BUF_BYTES);
    __bf16 (*cB)[LDSK] = (__bf16(*)[LDSK])(smem + cur * GEMM_BUF_BYTES + SA_BYTES);

#ifdef USE_TDM
    if (wave == 0) {
      if (kt + 1 < KT) {
        unsigned nb = (unsigned)((cur ^ 1) * GEMM_BUF_BYTES);
        tdm_load_2d(nb,            Abase + k0 + GBK, K, M, GBK, GBM, K, 3, 3);
        tdm_load_2d(nb + SA_BYTES, Bbase + k0 + GBK, K, N, GBK, GBN, K, 3, 3);
        __builtin_amdgcn_s_wait_tensorcnt(2);   // current buffer's 2 loads done
      } else {
        __builtin_amdgcn_s_wait_tensorcnt(0);
      }
      // TDM builtin carries no aliasing info about smem: force the compiler
      // to treat LDS as written so fragment ds_loads are not folded away.
      asm volatile("" ::: "memory");
    }
    __syncthreads();
#else
    // manual staging: both tiles are 128 rows x 32 k-contiguous bf16
    #pragma unroll
    for (int i = 0; i < 2; ++i) {
      int c = i * 256 + tid;
      int r = c >> 2, kc = (c & 3) * 8;
      *(uint4*)&cA[r][kc] = *(const uint4*)&Abase[(size_t)r * K + k0 + kc];
      *(uint4*)&cB[r][kc] = *(const uint4*)&Bbase[(size_t)r * K + k0 + kc];
    }
    if (k0 + GBK < K) {
      __builtin_prefetch(&Abase[(size_t)(tid >> 1) * K + k0 + GBK], 0, 0);
      __builtin_prefetch(&Bbase[(size_t)(tid >> 1) * K + k0 + GBK], 0, 0);
    }
    __syncthreads();
#endif

    v16bf afr[2], bfr[4];
    #pragma unroll
    for (int mc = 0; mc < 2; ++mc) {
      int r = wm * 32 + mc * 16 + lr;
      #pragma unroll
      for (int j = 0; j < 16; ++j) afr[mc][j] = cA[r][lh + j];
    }
    #pragma unroll
    for (int nc = 0; nc < 4; ++nc) {
      int n = wn * 64 + nc * 16 + lr;
      #pragma unroll
      for (int j = 0; j < 16; ++j) bfr[nc][j] = cB[n][lh + j];
    }
    #pragma unroll
    for (int mc = 0; mc < 2; ++mc)
      #pragma unroll
      for (int nc = 0; nc < 4; ++nc)
        acc[mc][nc] = wmma_bf16(afr[mc], bfr[nc], acc[mc][nc]);
    __syncthreads();   // all waves done with cur before it is refilled
  }

  // epilogue
  const int mofs = (lane >= 16) ? 8 : 0;
  #pragma unroll
  for (int mc = 0; mc < 2; ++mc) {
    #pragma unroll
    for (int nc = 0; nc < 4; ++nc) {
      #pragma unroll
      for (int r = 0; r < 8; ++r) {
        int m = row0 + mc * 16 + r + mofs;   // global row (token)
        int c = col0 + nc * 16 + lr;         // global col
        float v = acc[mc][nc][r] + bias[c];
        if constexpr (MODE == 0) {
          int which = c >> 10;
          int cc = c & 1023;
          int hh = cc >> 6, dd = cc & 63;
          int b = m >> 10, t = m & 1023;
          int bh = b * NHEAD + hh;
          if (which == 0) {
            Qbf[((size_t)bh * TC + t) * HEAD_DIM + dd] = (__bf16)v;
          } else if (which == 1) {
            size_t o = ((size_t)bh * TKMAX + TP + t) * HEAD_DIM + dd;
            Koutf[o] = v; Kbf[o] = (__bf16)v;
          } else {
            size_t o = ((size_t)bh * TKMAX + TP + t) * HEAD_DIM + dd;
            Voutf[o] = v;
            Vtbf[((size_t)bh * HEAD_DIM + dd) * TKMAX + TP + t] = (__bf16)v;
          }
        } else if constexpr (MODE == 1) {
          outf[(size_t)m * N + c] = v + resid[(size_t)m * N + c];
        } else {
          float ge = 0.5f * v * (1.f + erff(v * 0.70710678118654752f));
          outbf[(size_t)m * N + c] = (__bf16)ge;
        }
      }
    }
  }
}

// ---------------------------------------------------------------------------
// Flash-style causal attention with WMMA (bf16 in, f32 accumulate)
// grid: (B*H*(TC/128)) blocks, 256 threads = 8 waves.
// wave w handles 16 query rows; key tiles of 32, double-buffered TDM staging.
// ---------------------------------------------------------------------------
#define SK_BYTES  (32 * 72 * 2)                     // 4608  [key][d]
#define SVT_BYTES (64 * 40 * 2)                     // 5120  [d][key]
#define KV_BUF_BYTES (SK_BYTES + SVT_BYTES)         // 9728 per buffer
#define SP_OFF   (2 * KV_BUF_BYTES)                 // 19456
#define SP_BYTES (8 * 16 * 36 * 2)                  // 9216
#define ATT_SMEM (SP_OFF + SP_BYTES)                // 28672

__global__ __launch_bounds__(256) void k_attention(
    const __bf16* __restrict__ Qbf, const __bf16* __restrict__ Kbf,
    const __bf16* __restrict__ Vtbf, const int* __restrict__ pos_ptr,
    __bf16* __restrict__ Obf) {
  __shared__ __align__(16) char smem[ATT_SMEM];
  __bf16 (*sP)[16][36] = (__bf16(*)[16][36])(smem + SP_OFF);

  const int tid = threadIdx.x;
  const int wave = tid >> 5, lane = tid & 31;
  const int qt = blockIdx.x & 7;           // query tile of 128
  const int bh = blockIdx.x >> 3;          // b*NHEAD + h
  const int pos = pos_ptr[0];
  const int qbase = qt * 128;
  const int qrow = qbase + wave * 16;
  const int lr = lane & 15;
  const int lh = (lane >> 4) * 16;
  const int mofs = (lane >= 16) ? 8 : 0;

  const __bf16* Qp = Qbf + ((size_t)bh * TC + qrow) * HEAD_DIM;
  const __bf16* Kb = Kbf + (size_t)bh * TKMAX * HEAD_DIM;
  const __bf16* Vb = Vtbf + (size_t)bh * HEAD_DIM * TKMAX;

  // load Q fragments (16x64 -> two 16x32 A-fragments) with explicit 16B loads
  v16bf qf[2];
  #pragma unroll
  for (int f = 0; f < 2; ++f) {
    union { uint4 u[2]; v16bf v; } tmp;
    const uint4* p = (const uint4*)(Qp + (size_t)lr * HEAD_DIM + f * 32 + lh);
    tmp.u[0] = p[0];
    tmp.u[1] = p[1];
    qf[f] = tmp.v;
  }

  float mrow[8], lrow[8];
  v8f o[4];
  #pragma unroll
  for (int r = 0; r < 8; ++r) { mrow[r] = -1e30f; lrow[r] = 0.f; }
  #pragma unroll
  for (int d = 0; d < 4; ++d) o[d] = (v8f)(0.f);

  const float scale = 0.125f;  // 1/sqrt(64)
  const int kmax = pos + qbase + 128;
  const int ntiles = (kmax + 31) / 32;

#ifdef USE_TDM
  if (wave == 0) {
    tdm_load_2d(0,        Kb, HEAD_DIM, TKMAX, HEAD_DIM, 32, HEAD_DIM, 4, 3);
    tdm_load_2d(SK_BYTES, Vb, TKMAX, HEAD_DIM, 32, HEAD_DIM, TKMAX, 3, 3);
  }
#endif

  for (int kt = 0; kt < ntiles; ++kt) {
    const int kofs = kt * 32;
    const int cur = kt & 1;
    __bf16 (*sK)[72]  = (__bf16(*)[72])(smem + cur * KV_BUF_BYTES);
    __bf16 (*sVt)[40] = (__bf16(*)[40])(smem + cur * KV_BUF_BYTES + SK_BYTES);

#ifdef USE_TDM
    if (wave == 0) {
      if (kt + 1 < ntiles) {
        unsigned nb = (unsigned)((cur ^ 1) * KV_BUF_BYTES);
        tdm_load_2d(nb, Kb + (size_t)(kofs + 32) * HEAD_DIM,
                    HEAD_DIM, TKMAX, HEAD_DIM, 32, HEAD_DIM, 4, 3);
        tdm_load_2d(nb + SK_BYTES, Vb + kofs + 32,
                    TKMAX, HEAD_DIM, 32, HEAD_DIM, TKMAX, 3, 3);
        __builtin_amdgcn_s_wait_tensorcnt(2);
      } else {
        __builtin_amdgcn_s_wait_tensorcnt(0);
      }
      asm volatile("" ::: "memory");   // see GEMM: make TDM writes visible
    }
    __syncthreads();
#else
    {
      int r = tid >> 3, c = (tid & 7) * 8;
      *(uint4*)&sK[r][c] = *(const uint4*)&Kb[(size_t)(kofs + r) * HEAD_DIM + c];
      int d = tid >> 2, kc = (tid & 3) * 8;
      *(uint4*)&sVt[d][kc] = *(const uint4*)&Vb[(size_t)d * TKMAX + kofs + kc];
    }
    __syncthreads();
#endif

    // S = Q * K^T
    v8f s[2];
    #pragma unroll
    for (int nc = 0; nc < 2; ++nc) {
      s[nc] = (v8f)(0.f);
      v16bf bk0, bk1;
      int key = nc * 16 + lr;
      #pragma unroll
      for (int j = 0; j < 16; ++j) {
        bk0[j] = sK[key][lh + j];
        bk1[j] = sK[key][32 + lh + j];
      }
      s[nc] = wmma_bf16(qf[0], bk0, s[nc]);
      s[nc] = wmma_bf16(qf[1], bk1, s[nc]);
    }

    // online softmax per row; emit P (bf16) to LDS
    #pragma unroll
    for (int r = 0; r < 8; ++r) {
      int m  = r + mofs;
      int gi = qrow + m;
      float x0 = s[0][r] * scale;
      float x1 = s[1][r] * scale;
      if (kofs + lr > pos + gi)      x0 = -1e30f;
      if (kofs + 16 + lr > pos + gi) x1 = -1e30f;
      float mx = fmaxf(x0, x1);
      #pragma unroll
      for (int off = 8; off >= 1; off >>= 1) mx = fmaxf(mx, __shfl_xor(mx, off, 32));
      float mnew = fmaxf(mrow[r], mx);
      float corr = __expf(mrow[r] - mnew);
      float p0 = __expf(x0 - mnew);
      float p1 = __expf(x1 - mnew);
      float sum = p0 + p1;
      #pragma unroll
      for (int off = 8; off >= 1; off >>= 1) sum += __shfl_xor(sum, off, 32);
      lrow[r] = lrow[r] * corr + sum;
      mrow[r] = mnew;
      #pragma unroll
      for (int d = 0; d < 4; ++d) o[d][r] *= corr;
      sP[wave][m][lr]      = (__bf16)p0;
      sP[wave][m][16 + lr] = (__bf16)p1;
    }
    __syncthreads();

    // O += P(16x32) * V(32x64)
    v16bf pf;
    #pragma unroll
    for (int j = 0; j < 16; ++j) pf[j] = sP[wave][lr][lh + j];
    #pragma unroll
    for (int dc = 0; dc < 4; ++dc) {
      v16bf bv;
      int d = dc * 16 + lr;
      #pragma unroll
      for (int j = 0; j < 16; ++j) bv[j] = sVt[d][lh + j];
      o[dc] = wmma_bf16(pf, bv, o[dc]);
    }
    __syncthreads();   // done reading cur buffer before refill
  }

  // normalize + write O into [token, d_model] bf16 buffer at head column
  const int b = bh >> 4, h = bh & 15;
  #pragma unroll
  for (int dc = 0; dc < 4; ++dc) {
    #pragma unroll
    for (int r = 0; r < 8; ++r) {
      int m = r + mofs;
      int tok = b * TC + qrow + m;
      int col = h * HEAD_DIM + dc * 16 + lr;
      Obf[(size_t)tok * D_MODEL + col] = (__bf16)(o[dc][r] / lrow[r]);
    }
  }
}

// ---------------------------------------------------------------------------
// Host-side launch
// ---------------------------------------------------------------------------
extern "C" void kernel_launch(void* const* d_in, const int* in_sizes, int n_in,
                              void* d_out, int out_size, void* d_ws, size_t ws_size,
                              hipStream_t stream) {
  const float* x    = (const float*)d_in[0];
  const float* Kp   = (const float*)d_in[1];
  const float* Vp   = (const float*)d_in[2];
  const int*   pos  = (const int*)d_in[3];
  const float* wqkv = (const float*)d_in[4];
  const float* bqkv = (const float*)d_in[5];
  const float* wo   = (const float*)d_in[6];
  const float* bo   = (const float*)d_in[7];
  const float* g1   = (const float*)d_in[8];
  const float* be1  = (const float*)d_in[9];
  const float* g2   = (const float*)d_in[10];
  const float* be2  = (const float*)d_in[11];
  const float* w1   = (const float*)d_in[12];
  const float* bf1  = (const float*)d_in[13];
  const float* w2   = (const float*)d_in[14];
  const float* bf2  = (const float*)d_in[15];

  float* xout = (float*)d_out;
  float* Kout = xout + (size_t)NTOK * D_MODEL;
  float* Vout = Kout + (size_t)BATCH * NHEAD * TKMAX * HEAD_DIM;

  char* w = (char*)d_ws;
  size_t off = 0;
  auto carve = [&](size_t bytes) {
    void* p = w + off;
    off += (bytes + 255) & ~(size_t)255;
    return p;
  };
  // weights stored TRANSPOSED [N][K] in bf16
  __bf16* wqkvT = (__bf16*)carve((size_t)3 * D_MODEL * D_MODEL * 2);
  __bf16* woT   = (__bf16*)carve((size_t)D_MODEL * D_MODEL * 2);
  __bf16* w1T   = (__bf16*)carve((size_t)DIM_FF * D_MODEL * 2);
  __bf16* w2T   = (__bf16*)carve((size_t)D_MODEL * DIM_FF * 2);
  __bf16* h_bf  = (__bf16*)carve((size_t)NTOK * D_MODEL * 2);    // reused for h2
  __bf16* q_bf  = (__bf16*)carve((size_t)NTOK * D_MODEL * 2);
  __bf16* K_bf  = (__bf16*)carve((size_t)BATCH * NHEAD * TKMAX * HEAD_DIM * 2);
  __bf16* Vt_bf = (__bf16*)carve((size_t)BATCH * NHEAD * HEAD_DIM * TKMAX * 2);
  __bf16* o_bf  = (__bf16*)carve((size_t)NTOK * D_MODEL * 2);
  float*  x1    = (float*)carve((size_t)NTOK * D_MODEL * 4);
  __bf16* ff_bf = K_bf;  // FFN activation aliases dead bf16 KV shadow (32MB)

  // 1) weight converts (transpose to [N][K])
  {
    int n;
    n = D_MODEL * 3 * D_MODEL; k_cvt_transpose<<<(n + 255) / 256, 256, 0, stream>>>(wqkv, wqkvT, D_MODEL, 3 * D_MODEL);
    n = D_MODEL * D_MODEL;     k_cvt_transpose<<<(n + 255) / 256, 256, 0, stream>>>(wo, woT, D_MODEL, D_MODEL);
    n = D_MODEL * DIM_FF;      k_cvt_transpose<<<(n + 255) / 256, 256, 0, stream>>>(w1, w1T, D_MODEL, DIM_FF);
    n = DIM_FF * D_MODEL;      k_cvt_transpose<<<(n + 255) / 256, 256, 0, stream>>>(w2, w2T, DIM_FF, D_MODEL);
  }

  // 2) LN1
  k_layernorm<<<NTOK, 256, 0, stream>>>(x, g1, be1, h_bf);

  // 3) past KV copy into cache + bf16 shadows
  {
    int n = BATCH * NHEAD * TP * HEAD_DIM;
    k_kvcopy<<<(n + 255) / 256, 256, 0, stream>>>(Kp, Vp, Kout, Vout, K_bf, Vt_bf);
  }

  // 4) QKV GEMM: [4096,1024] x [1024,3072]
  {
    int M = NTOK, N = 3 * D_MODEL, K = D_MODEL;
    dim3 grid((M / GBM) * (N / GBN));
    k_gemm_bf16<0><<<grid, 256, 0, stream>>>(h_bf, wqkvT, M, N, K, bqkv,
                                             Kout, Vout, K_bf, Vt_bf, q_bf,
                                             nullptr, nullptr, nullptr);
  }

  // 5) attention
  k_attention<<<BATCH * NHEAD * (TC / 128), 256, 0, stream>>>(q_bf, K_bf, Vt_bf, pos, o_bf);

  // 6) Wo GEMM + residual -> x1
  {
    int M = NTOK, N = D_MODEL, K = D_MODEL;
    dim3 grid((M / GBM) * (N / GBN));
    k_gemm_bf16<1><<<grid, 256, 0, stream>>>(o_bf, woT, M, N, K, bo,
                                             nullptr, nullptr, nullptr, nullptr, nullptr,
                                             x, x1, nullptr);
  }

  // 7) LN2 -> h2 (reuse h_bf)
  k_layernorm<<<NTOK, 256, 0, stream>>>(x1, g2, be2, h_bf);

  // 8) W1 GEMM + GELU -> ff_bf
  {
    int M = NTOK, N = DIM_FF, K = D_MODEL;
    dim3 grid((M / GBM) * (N / GBN));
    k_gemm_bf16<2><<<grid, 256, 0, stream>>>(h_bf, w1T, M, N, K, bf1,
                                             nullptr, nullptr, nullptr, nullptr, nullptr,
                                             nullptr, nullptr, ff_bf);
  }

  // 9) W2 GEMM + residual -> xout
  {
    int M = NTOK, N = D_MODEL, K = DIM_FF;
    dim3 grid((M / GBM) * (N / GBN));
    k_gemm_bf16<1><<<grid, 256, 0, stream>>>(ff_bf, w2T, M, N, K, bf2,
                                             nullptr, nullptr, nullptr, nullptr, nullptr,
                                             x1, xout, nullptr);
  }
}
